// MultiHeadAttention_32263794327915
// MI455X (gfx1250) — matrix operand order repaired and checked by reading, not compiled, not run on hardware
//
#include <hip/hip_runtime.h>

// ---------------------------------------------------------------------------
// MHA forward for MI455X (gfx1250): bf16 WMMA with fp32 accumulate,
// async-to-LDS double-buffered K/V staging in flash attention.
// B=2, S=2048, D=1024, H=16, dh=64.
// ---------------------------------------------------------------------------

typedef __bf16 bf16_t;
typedef __attribute__((ext_vector_type(16))) __bf16 v16bf;
typedef __attribute__((ext_vector_type(8)))  __bf16 v8bf;
typedef __attribute__((ext_vector_type(4)))  __bf16 v4bf;
typedef __attribute__((ext_vector_type(8)))  float  v8f;
typedef __attribute__((ext_vector_type(4)))  int    v4i;

#define BATCH 2
#define SEQ   2048
#define DMODEL 1024
#define NH    16
#define DH    64
#define MTOT  (BATCH*SEQ)     // 4096 rows

#define LDS_AS __attribute__((address_space(3)))

union Frag { v16bf v; v8bf h[2]; };

static __device__ __host__ inline bf16_t f2bf(float f) {
  union { float f; unsigned u; } x; x.f = f;
  unsigned r = x.u + 0x7FFFu + ((x.u >> 16) & 1u);   // round-to-nearest-even
  unsigned short hh = (unsigned short)(r >> 16);
  return __builtin_bit_cast(bf16_t, hh);
}

// --- CDNA5 async copy: 16 bytes global -> LDS per lane (ASYNCcnt-tracked) ---
static __device__ inline void async_copy16(const bf16_t* g, bf16_t* l) {
#if __has_builtin(__builtin_amdgcn_global_load_async_to_lds_b128)
  __builtin_amdgcn_global_load_async_to_lds_b128((v4i*)g, (v4i*)l, 0, 0);
#else
  unsigned loff = (unsigned)(size_t)(LDS_AS bf16_t*)l;
  asm volatile("global_load_async_to_lds_b128 %0, %1, off"
               :: "v"(loff), "v"((unsigned long long)g) : "memory");
#endif
}

static __device__ inline void async_wait0() {
#if __has_builtin(__builtin_amdgcn_s_wait_asynccnt)
  __builtin_amdgcn_s_wait_asynccnt(0);
#else
  asm volatile("s_wait_asynccnt 0x0" ::: "memory");
#endif
}

// ---------------------------------------------------------------------------
// fp32 -> bf16 conversion, vectorized (float4 in, 8B bf16x4 out)
// ---------------------------------------------------------------------------
__global__ __launch_bounds__(256)
void cvt_f32_bf16(const float4* __restrict__ src, bf16_t* __restrict__ dst, int n4) {
  int i = blockIdx.x * blockDim.x + threadIdx.x;
  int stride = gridDim.x * blockDim.x;
  for (; i < n4; i += stride) {
    float4 f = src[i];
    v4bf o;
    o[0] = f2bf(f.x); o[1] = f2bf(f.y); o[2] = f2bf(f.z); o[3] = f2bf(f.w);
    *(v4bf*)(dst + 4 * (size_t)i) = o;
  }
}

// ---------------------------------------------------------------------------
// GEMM: C[m,n] = sum_k A[m,k] * W[n,k] + bias[n]   (nn.Linear: x @ W.T + b)
// A: [M,K] bf16 row-major, W: [N,K] bf16 row-major.
// Wave computes 32Mx64N (8 WMMAs / K-step), block = 8 waves = 256Mx64N.
// Register double-buffered over K so loads overlap WMMAs.
// mode 0: fp32 row-major [M,N] out
// mode 1: bf16 [B,H,S,dh] out, scaled (Q)
// mode 2: bf16 [B,H,S,dh] out (K)
// mode 3: bf16 [B,H,dh,S] out (V transposed)
// ---------------------------------------------------------------------------
__global__ __launch_bounds__(256)
void gemm_bf16(const bf16_t* __restrict__ A, const bf16_t* __restrict__ W,
               const float* __restrict__ bias, void* __restrict__ out,
               int M, int N, int K, int mode, float scale) {
  const int lane = threadIdx.x & 31;
  const int wid  = threadIdx.x >> 5;
  const int ln   = lane & 15;
  const int half = lane >> 4;

  const int m0 = blockIdx.y * 256 + wid * 32;
  const int n0 = blockIdx.x * 64;

  v8f zero = {};
  v8f acc[2][4] = {{zero, zero, zero, zero}, {zero, zero, zero, zero}};

  // A-operand per-lane K chunks (ISA 16-bit A layout):
  //   lanes 0-15: K = {0..7, 16..23}; lanes 16-31: K = {8..15, 24..31}
  const int offA0 = half * 8;
  const int offA1 = 16 + half * 8;
  // B-operand per-lane K chunks (same order as A):
  //   lanes 0-15: K = {0..7, 16..23}; lanes 16-31: K = {8..15, 24..31}
  const int offB  = half * 8;

  const bf16_t* Arow0 = A + (size_t)(m0 + ln) * K;
  const bf16_t* Arow1 = A + (size_t)(m0 + 16 + ln) * K;

  auto loadA = [&](Frag* d, int kk) {
    d[0].h[0] = *(const v8bf*)(Arow0 + kk + offA0);
    d[0].h[1] = *(const v8bf*)(Arow0 + kk + offA1);
    d[1].h[0] = *(const v8bf*)(Arow1 + kk + offA0);
    d[1].h[1] = *(const v8bf*)(Arow1 + kk + offA1);
  };
  auto loadB = [&](Frag* d, int kk) {
#pragma unroll
    for (int j = 0; j < 4; ++j) {
      const bf16_t* Wrow = W + (size_t)(n0 + j * 16 + ln) * K + kk;
      d[j].h[0] = *(const v8bf*)(Wrow + offB);
      d[j].h[1] = *(const v8bf*)(Wrow + offB + 16);
    }
  };
  auto step = [&](Frag* a, Frag* b) {
#pragma unroll
    for (int mi = 0; mi < 2; ++mi)
#pragma unroll
      for (int j = 0; j < 4; ++j)
        acc[mi][j] = __builtin_amdgcn_wmma_f32_16x16x32_bf16(
            false, a[mi].v, false, b[j].v, (short)0, acc[mi][j], false, false);
  };

  Frag aC[2], bC[4], aN[2], bN[4];
  loadA(aC, 0); loadB(bC, 0);
  for (int kk = 0; kk < K; kk += 64) {        // K multiple of 64
    loadA(aN, kk + 32); loadB(bN, kk + 32);
    if (kk + 128 < K) __builtin_prefetch(Arow0 + kk + 128, 0, 0);
    step(aC, bC);
    if (kk + 64 < K) { loadA(aC, kk + 64); loadB(bC, kk + 64); }
    step(aN, bN);
  }

  // Store. D-layout: VGPR r, lanes 0-15 -> row base+r, lanes 16-31 -> base+8+r.
  if (mode == 0) {
    float* outF = (float*)out;
#pragma unroll
    for (int mi = 0; mi < 2; ++mi)
#pragma unroll
    for (int j = 0; j < 4; ++j) {
      const int col = n0 + j * 16 + ln;
      const float bv = bias[col];
#pragma unroll
      for (int r = 0; r < 8; ++r) {
        const int row = m0 + mi * 16 + r + 8 * half;
        outF[(size_t)row * N + col] = acc[mi][j][r] + bv;
      }
    }
  } else {
    bf16_t* outB = (bf16_t*)out;
#pragma unroll
    for (int mi = 0; mi < 2; ++mi)
#pragma unroll
    for (int j = 0; j < 4; ++j) {
      const int col = n0 + j * 16 + ln;
      const float bv = bias[col];
      const int hh = col >> 6;      // head
      const int d  = col & 63;      // within-head dim
#pragma unroll
      for (int r = 0; r < 8; ++r) {
        const int row = m0 + mi * 16 + r + 8 * half;
        const int bb = row >> 11;   // batch (S = 2048)
        const int s  = row & 2047;
        const float v = (acc[mi][j][r] + bv) * scale;
        size_t idx;
        if (mode == 3) idx = (((size_t)bb * NH + hh) * DH + d) * SEQ + s;   // V^T
        else           idx = (((size_t)bb * NH + hh) * SEQ + s) * DH + d;   // Q,K
        outB[idx] = f2bf(v);
      }
    }
  }
}

// ---------------------------------------------------------------------------
// Flash attention: one block per (b, h, 128-query slab); 8 waves, each owns a
// 16-query tile. K/V 32-key chunks are staged cooperatively into
// double-buffered LDS with global_load_async_to_lds_b128 and shared by all
// 8 waves; score->P re-layout goes through a per-wave LDS tile.
// Q,K: [B,H,S,dh] bf16 (Q pre-scaled by 1/sqrt(dh)); Vt: [B,H,dh,S] bf16.
// ctx out: [B,S,D] bf16.
// ---------------------------------------------------------------------------
__global__ __launch_bounds__(256)
void flash_attn(const bf16_t* __restrict__ Q, const bf16_t* __restrict__ K,
                const bf16_t* __restrict__ Vt, bf16_t* __restrict__ ctx) {
  __shared__ __attribute__((aligned(16))) bf16_t ldsK[2][32 * 64]; // [key][dh]
  __shared__ __attribute__((aligned(16))) bf16_t ldsV[2][64 * 32]; // [dh][key]
  __shared__ __attribute__((aligned(16))) bf16_t ldsP[8][16 * 32]; // per-wave P

  const int tid  = threadIdx.x;
  const int lane = tid & 31;
  const int wid  = tid >> 5;
  const int ln   = lane & 15;
  const int half = lane >> 4;

  const int nQB = SEQ / 128;                     // 16 slabs per (b,h)
  const int gid = blockIdx.x;                    // 0..B*NH*nQB-1 = 511
  const int b   = gid / (NH * nQB);
  const int rem = gid % (NH * nQB);
  const int h   = rem / nQB;
  const int qb  = rem % nQB;
  const int q0  = qb * 128 + wid * 16;           // this wave's query tile

  const bf16_t* Qb = Q  + ((size_t)b * NH + h) * SEQ * DH;
  const bf16_t* Kb = K  + ((size_t)b * NH + h) * SEQ * DH;
  const bf16_t* Vb = Vt + ((size_t)b * NH + h) * DH * SEQ;

  // Per-thread async-copy coordinates (each thread moves 16B per tile).
  const int krow = tid >> 3, kcol = (tid & 7) * 8;   // K tile: 32 x 64
  const int vrow = tid >> 2, vcol = (tid & 3) * 8;   // V tile: 64 x 32

  // Load the 16x64 Q tile once as two K=32 A-fragments.
  Frag aQ[2];
  {
    const bf16_t* qrow = Qb + (size_t)(q0 + ln) * DH;
#pragma unroll
    for (int f = 0; f < 2; ++f) {
      aQ[f].h[0] = *(const v8bf*)(qrow + f * 32 + half * 8);
      aQ[f].h[1] = *(const v8bf*)(qrow + f * 32 + 16 + half * 8);
    }
  }

  v8f zero = {};
  v8f accO[4] = {zero, zero, zero, zero};        // 16 x 64 output accumulator
  float rowm[8], rowl[8];
#pragma unroll
  for (int r = 0; r < 8; ++r) { rowm[r] = -3.0e30f; rowl[r] = 0.0f; }

  const int myKend = q0 + 16;                    // causal bound for this wave
  const int nch    = (qb * 128 + 128) / 32;      // chunks for whole slab

  auto stage = [&](int c) {
    const int kb = c * 32, bufi = c & 1;
    async_copy16(Kb + (size_t)(kb + krow) * DH + kcol, &ldsK[bufi][krow * 64 + kcol]);
    async_copy16(Vb + (size_t)vrow * SEQ + kb + vcol,  &ldsV[bufi][vrow * 32 + vcol]);
  };

  stage(0);
  async_wait0();
  __syncthreads();

  for (int c = 0; c < nch; ++c) {
    const int kb = c * 32, bufi = c & 1;
    if (c + 1 < nch) stage(c + 1);               // prefetch next chunk (async)

    if (kb < myKend) {
      // ---- scores: 16q x 32k as two 16x16 tiles, K-dim = dh = 2 frags ----
      v8f s[2];
#pragma unroll
      for (int t = 0; t < 2; ++t) {
        const bf16_t* krowp = &ldsK[bufi][(t * 16 + ln) * 64];
        v8f cc = {};
#pragma unroll
        for (int f = 0; f < 2; ++f) {
          Frag bK;
          bK.h[0] = *(const v8bf*)(krowp + f * 32 + half * 8);
          bK.h[1] = *(const v8bf*)(krowp + f * 32 + 16 + half * 8);
          cc = __builtin_amdgcn_wmma_f32_16x16x32_bf16(
              false, aQ[f].v, false, bK.v, (short)0, cc, false, false);
        }
        s[t] = cc;
      }

      // ---- causal mask in D-layout ----
#pragma unroll
      for (int t = 0; t < 2; ++t) {
        const int key = kb + t * 16 + ln;
#pragma unroll
        for (int r = 0; r < 8; ++r) {
          const int qrow = q0 + r + 8 * half;
          if (key > qrow) s[t][r] = -3.0e30f;
        }
      }

      // ---- online softmax: row max via 16-lane butterfly ----
      float alpha[8];
#pragma unroll
      for (int r = 0; r < 8; ++r) {
        float m = fmaxf(s[0][r], s[1][r]);
#pragma unroll
        for (int sh = 1; sh < 16; sh <<= 1) m = fmaxf(m, __shfl_xor(m, sh, 32));
        const float mnew = fmaxf(rowm[r], m);
        alpha[r] = __expf(rowm[r] - mnew);
        rowm[r] = mnew;
      }

      // ---- P = exp(s - m): row sums + spill to LDS for A-layout reload ----
#pragma unroll
      for (int t = 0; t < 2; ++t) {
#pragma unroll
        for (int r = 0; r < 8; ++r) {
          const float p = __expf(s[t][r] - rowm[r]);
          s[t][r] = p;
          ldsP[wid][(r + 8 * half) * 32 + t * 16 + ln] = f2bf(p);
        }
      }
#pragma unroll
      for (int r = 0; r < 8; ++r) {
        float sum = s[0][r] + s[1][r];
#pragma unroll
        for (int sh = 1; sh < 16; sh <<= 1) sum += __shfl_xor(sum, sh, 32);
        rowl[r] = rowl[r] * alpha[r] + sum;
      }

      // ---- rescale accumulator ----
#pragma unroll
      for (int j = 0; j < 4; ++j)
#pragma unroll
        for (int r = 0; r < 8; ++r) accO[j][r] *= alpha[r];

      // ---- reload P as A-operand (16x32, K = keys) ----
      Frag aP;
      const bf16_t* prow = &ldsP[wid][ln * 32];
      aP.h[0] = *(const v8bf*)(prow + half * 8);
      aP.h[1] = *(const v8bf*)(prow + 16 + half * 8);

      // ---- accO += P @ V  from the shared V^T LDS tile ----
#pragma unroll
      for (int j = 0; j < 4; ++j) {
        Frag bV;
        const bf16_t* vrowp = &ldsV[bufi][(j * 16 + ln) * 32];
        bV.h[0] = *(const v8bf*)(vrowp + half * 8);
        bV.h[1] = *(const v8bf*)(vrowp + 16 + half * 8);
        accO[j] = __builtin_amdgcn_wmma_f32_16x16x32_bf16(
            false, aP.v, false, bV.v, (short)0, accO[j], false, false);
      }
    }

    async_wait0();       // our async writes into buf[(c+1)&1] are done
    __syncthreads();     // everyone finished reading buf[c&1] + sees new data
  }

  // ---- normalize and store ctx[b, s, h*64 + d] ----
#pragma unroll
  for (int j = 0; j < 4; ++j) {
#pragma unroll
    for (int r = 0; r < 8; ++r) {
      const float o = accO[j][r] / rowl[r];
      const int srow = q0 + r + 8 * half;
      const size_t idx = ((size_t)b * SEQ + srow) * DMODEL + h * DH + j * 16 + ln;
      ctx[idx] = f2bf(o);
    }
  }
}

// ---------------------------------------------------------------------------
// Host launcher
// ---------------------------------------------------------------------------
extern "C" void kernel_launch(void* const* d_in, const int* in_sizes, int n_in,
                              void* d_out, int out_size, void* d_ws, size_t ws_size,
                              hipStream_t stream) {
  const float* hs = (const float*)d_in[0];
  const float* Wq = (const float*)d_in[1];
  const float* bq = (const float*)d_in[2];
  const float* Wk = (const float*)d_in[3];
  const float* bk = (const float*)d_in[4];
  const float* Wv = (const float*)d_in[5];
  const float* bv = (const float*)d_in[6];
  const float* Wo = (const float*)d_in[7];
  const float* bo = (const float*)d_in[8];
  float* out = (float*)d_out;

  char* ws = (char*)d_ws;
  size_t off = 0;
  const size_t xbytes = (size_t)MTOT * DMODEL * sizeof(bf16_t);       // 8 MB
  const size_t wbytes = (size_t)DMODEL * DMODEL * sizeof(bf16_t);     // 2 MB

  bf16_t* Xb   = (bf16_t*)(ws + off); off += xbytes;
  bf16_t* Wqb  = (bf16_t*)(ws + off); off += wbytes;
  bf16_t* Wkb  = (bf16_t*)(ws + off); off += wbytes;
  bf16_t* Wvb  = (bf16_t*)(ws + off); off += wbytes;
  bf16_t* Wob  = (bf16_t*)(ws + off); off += wbytes;
  bf16_t* Qbuf = (bf16_t*)(ws + off); off += xbytes;
  bf16_t* Kbuf = (bf16_t*)(ws + off); off += xbytes;
  bf16_t* Vbuf = (bf16_t*)(ws + off); off += xbytes;                  // transposed
  bf16_t* Ctx  = (bf16_t*)(ws + off); off += xbytes;                  // total 48 MB

  const int nX4 = MTOT * DMODEL / 4;
  const int nW4 = DMODEL * DMODEL / 4;
  cvt_f32_bf16<<<1024, 256, 0, stream>>>((const float4*)hs, Xb, nX4);
  cvt_f32_bf16<<<512, 256, 0, stream>>>((const float4*)Wq, Wqb, nW4);
  cvt_f32_bf16<<<512, 256, 0, stream>>>((const float4*)Wk, Wkb, nW4);
  cvt_f32_bf16<<<512, 256, 0, stream>>>((const float4*)Wv, Wvb, nW4);
  cvt_f32_bf16<<<512, 256, 0, stream>>>((const float4*)Wo, Wob, nW4);

  const dim3 ggrid(DMODEL / 64, MTOT / 256);     // (16, 16)
  const float qscale = 0.125f;                   // 1/sqrt(dh), dh = 64
  gemm_bf16<<<ggrid, 256, 0, stream>>>(Xb, Wqb, bq, Qbuf, MTOT, DMODEL, DMODEL, 1, qscale);
  gemm_bf16<<<ggrid, 256, 0, stream>>>(Xb, Wkb, bk, Kbuf, MTOT, DMODEL, DMODEL, 2, 1.0f);
  gemm_bf16<<<ggrid, 256, 0, stream>>>(Xb, Wvb, bv, Vbuf, MTOT, DMODEL, DMODEL, 3, 1.0f);

  flash_attn<<<BATCH * NH * (SEQ / 128), 256, 0, stream>>>(Qbuf, Kbuf, Vbuf, Ctx);

  gemm_bf16<<<ggrid, 256, 0, stream>>>(Ctx, Wob, bo, out, MTOT, DMODEL, DMODEL, 0, 1.0f);
}
